// MaskedAttention_21191368639160
// MI455X (gfx1250) — compile-verified, matrix operand order checked by reading, not run on hardware
//
#include <hip/hip_runtime.h>

typedef unsigned short u16;
typedef __attribute__((ext_vector_type(16))) __bf16 v16bf;
typedef __attribute__((ext_vector_type(8)))  float  v8f;
typedef __attribute__((ext_vector_type(4)))  unsigned int uint4v;

union Frag {
  uint4v q[2];   // 2 x 16 bytes
  v16bf  v;      // 16 bf16
};

__device__ __forceinline__ u16 f2bf(float x) {
  unsigned u = __float_as_uint(x);
  unsigned r = u + 0x7FFFu + ((u >> 16) & 1u);   // round-to-nearest-even
  return (u16)(r >> 16);
}

__device__ __forceinline__ v8f wmma_bf16(v16bf a, v16bf b, v8f c) {
  return __builtin_amdgcn_wmma_f32_16x16x32_bf16(
      /*neg_a=*/false, a, /*neg_b=*/false, b,
      /*c_mod=*/(short)0, c, /*reuse_a=*/false, /*reuse_b=*/false);
}

// 16x32 bf16 A-fragment (B-fragment is the mirror: lane = column, read from the
// K-transposed matrix). CDNA5 16-bit A layout:
//   lane<16 : row = lane,    elems 0..7 = K k0..+7,   elems 8..15 = K k0+16..+23
//   lane>=16: row = lane-16, elems 0..7 = K k0+8..+15, elems 8..15 = K k0+24..+31
__device__ __forceinline__ v16bf load_frag(const u16* __restrict__ base, int ld,
                                           int row0, int k0, int lane) {
  const int r  = lane & 15;
  const int hi = lane >> 4;
  const u16* p = base + (size_t)(row0 + r) * ld + k0 + hi * 8;
  Frag f;
  f.q[0] = *(const uint4v*)(p);
  f.q[1] = *(const uint4v*)(p + 16);
  return f.v;
}

// ---------------------------------------------------------------- converts ---
__global__ void __launch_bounds__(256)
convert_bf16(const float* __restrict__ in, u16* __restrict__ out, int n) {
  int i = blockIdx.x * blockDim.x + threadIdx.x;
  if (i < n) out[i] = f2bf(in[i]);
}

// out[C][R] = bf16(in[R][C])
__global__ void __launch_bounds__(256)
transpose_f32_to_bf16(const float* __restrict__ in, u16* __restrict__ out,
                      int R, int C) {
  __shared__ float tile[32][33];
  const int bx = blockIdx.x * 32, by = blockIdx.y * 32;
  const int tx = threadIdx.x, ty = threadIdx.y;          // block 32 x 8
#pragma unroll
  for (int i = 0; i < 32; i += 8)
    tile[ty + i][tx] = in[(size_t)(by + ty + i) * C + bx + tx];
  __syncthreads();
#pragma unroll
  for (int i = 0; i < 32; i += 8)
    out[(size_t)(bx + ty + i) * R + by + tx] = f2bf(tile[tx][ty + i]);
}

// Bit-pack the 0/1 mask: one u32 word per 32 columns, built with wave32 ballot.
__global__ void __launch_bounds__(256)
pack_mask(const int* __restrict__ mask, unsigned* __restrict__ bits, int nWords) {
  const int i    = blockIdx.x * blockDim.x + threadIdx.x;
  const int word = i >> 5;
  const int lane = i & 31;
  if (word < nWords) {
    const int v = mask[(size_t)word * 32 + lane];
    const unsigned long long b = __ballot(v != 0);   // wave32: low 32 bits
    if (lane == 0) bits[word] = (unsigned)b;
  }
}

// ----------------------------------------------------------------- GEMMs ----
// One wave computes a 32x32 tile (2x2 WMMA register blocking) of
// A[rows x K] * Bt^T (+ bias).
// MODE 0: Q  -> [b,h,n,d] bf16
// MODE 1: KV -> K [b,h,m,d] bf16 and V^T [b,h,d,m] bf16
// MODE 2: f32 output (final projection)
template <int MODE>
__global__ void __launch_bounds__(128)
gemm_wmma(const u16* __restrict__ A, const u16* __restrict__ Bt,
          const float* __restrict__ bias,
          u16* __restrict__ O0, u16* __restrict__ O1, float* __restrict__ Of,
          int K, int cols) {
  const int lane = threadIdx.x & 31;
  const int wave = threadIdx.x >> 5;
  const int tile = blockIdx.x * 4 + wave;
  const int tilesN = cols >> 5;                 // 32-wide column tiles
  const int tM = (tile / tilesN) << 5;
  const int tN = (tile % tilesN) << 5;

  v8f acc[2][2];
#pragma unroll
  for (int i = 0; i < 2; ++i)
#pragma unroll
    for (int j = 0; j < 2; ++j)
#pragma unroll
      for (int e = 0; e < 8; ++e) acc[i][j][e] = 0.f;

  for (int k0 = 0; k0 < K; k0 += 32) {
    const v16bf a0 = load_frag(A,  K, tM,      k0, lane);
    const v16bf a1 = load_frag(A,  K, tM + 16, k0, lane);
    const v16bf b0 = load_frag(Bt, K, tN,      k0, lane);
    const v16bf b1 = load_frag(Bt, K, tN + 16, k0, lane);
    acc[0][0] = wmma_bf16(a0, b0, acc[0][0]);
    acc[0][1] = wmma_bf16(a0, b1, acc[0][1]);
    acc[1][0] = wmma_bf16(a1, b0, acc[1][0]);
    acc[1][1] = wmma_bf16(a1, b1, acc[1][1]);
  }

  const int hi = lane >> 4;
  const int nl = lane & 15;
#pragma unroll
  for (int j = 0; j < 2; ++j) {
    const int col = tN + j * 16 + nl;
    const float bv = bias[col];
#pragma unroll
    for (int i = 0; i < 2; ++i) {
#pragma unroll
      for (int r = 0; r < 8; ++r) {
        const int row = tM + i * 16 + r + 8 * hi;   // global row in [0,4096)
        const int bb  = row >> 10;
        const int n   = row & 1023;
        const float val = acc[i][j][r] + bv;
        if (MODE == 0) {
          const int h = col >> 6, dd = col & 63;
          O0[((((size_t)bb * 16 + h) << 10) + n) * 64 + dd] = f2bf(val);
        } else if (MODE == 1) {
          if (col < 1024) {
            const int h = col >> 6, dd = col & 63;
            O0[((((size_t)bb * 16 + h) << 10) + n) * 64 + dd] = f2bf(val);
          } else {
            const int c = col - 1024;
            const int h = c >> 6, dd = c & 63;
            O1[((((size_t)bb * 16 + h) * 64 + dd) << 10) + n] = f2bf(val);
          }
        } else {
          Of[(size_t)row * cols + col] = val;
        }
      }
    }
  }
}

// ------------------------------------------------------- flash attention ----
// One wave per (b, h, 16-row n-tile). Online softmax over m in 32-chunks.
__global__ void __launch_bounds__(128)
flash_attn(const u16* __restrict__ Q, const u16* __restrict__ Kb,
           const u16* __restrict__ Vt, const unsigned* __restrict__ mbits,
           u16* __restrict__ attn) {
  __shared__ __align__(16) u16 pls[4][16 * 40];   // probs tile, padded stride
  const int lane  = threadIdx.x & 31;
  const int wave  = threadIdx.x >> 5;
  const int wid   = blockIdx.x * 4 + wave;
  const int bh    = wid >> 6;             // b*16 + h
  const int nTile = (wid & 63) << 4;
  const int bb    = bh >> 4;
  const int h     = bh & 15;
  const int hi    = lane >> 4;
  const int nl    = lane & 15;

  const u16* Qh = Q  + (size_t)bh * 1024 * 64;
  const u16* Kh = Kb + (size_t)bh * 1024 * 64;
  const u16* Vh = Vt + (size_t)bh * 64 * 1024;

  const v16bf aQ0 = load_frag(Qh, 64, nTile, 0,  lane);
  const v16bf aQ1 = load_frag(Qh, 64, nTile, 32, lane);

  float mx[8], lsum[8];
  v8f o[4];
#pragma unroll
  for (int r = 0; r < 8; ++r) { mx[r] = -1e30f; lsum[r] = 0.f; }
#pragma unroll
  for (int t = 0; t < 4; ++t)
#pragma unroll
    for (int r = 0; r < 8; ++r) o[t][r] = 0.f;

  u16* lp = &pls[wave][0];

  for (int mc = 0; mc < 1024; mc += 32) {
    v8f z;
#pragma unroll
    for (int i = 0; i < 8; ++i) z[i] = 0.f;

    // S tile: 16 rows x 32 cols (two 16-col sub-tiles), K = d = 64
    v8f s0 = wmma_bf16(aQ0, load_frag(Kh, 64, mc,      0,  lane), z);
    s0     = wmma_bf16(aQ1, load_frag(Kh, 64, mc,      32, lane), s0);
    v8f s1 = wmma_bf16(aQ0, load_frag(Kh, 64, mc + 16, 0,  lane), z);
    s1     = wmma_bf16(aQ1, load_frag(Kh, 64, mc + 16, 32, lane), s1);

    // scale + mask from bit-packed words (bit==0 -> -inf surrogate)
#pragma unroll
    for (int r = 0; r < 8; ++r) {
      const int n = nTile + r + 8 * hi;
      const unsigned w = mbits[((size_t)bb * 1024 + n) * 32 + (mc >> 5)];
      s0[r] = ((w >> nl) & 1u)        ? s0[r] * 0.125f : -1e30f;
      s1[r] = ((w >> (nl + 16)) & 1u) ? s1[r] * 0.125f : -1e30f;
    }

    // online softmax: row reductions across the 16-lane halves (C layout)
    float f[8];
#pragma unroll
    for (int r = 0; r < 8; ++r) {
      float t = fmaxf(s0[r], s1[r]);
#pragma unroll
      for (int off = 8; off >= 1; off >>= 1) t = fmaxf(t, __shfl_xor(t, off, 32));
      const float nm = fmaxf(mx[r], t);
      f[r] = __expf(mx[r] - nm);
      mx[r] = nm;
      const float p0 = __expf(s0[r] - nm);
      const float p1 = __expf(s1[r] - nm);
      s0[r] = p0; s1[r] = p1;
      float rs = p0 + p1;
#pragma unroll
      for (int off = 8; off >= 1; off >>= 1) rs += __shfl_xor(rs, off, 32);
      lsum[r] = lsum[r] * f[r] + rs;
    }
#pragma unroll
    for (int t = 0; t < 4; ++t)
#pragma unroll
      for (int r = 0; r < 8; ++r) o[t][r] *= f[r];

    // stage P (bf16) through LDS to re-read in A-fragment layout
#pragma unroll
    for (int r = 0; r < 8; ++r) {
      const int row = r + 8 * hi;
      lp[row * 40 + nl]      = f2bf(s0[r]);
      lp[row * 40 + 16 + nl] = f2bf(s1[r]);
    }
    asm volatile("s_wait_dscnt 0" ::: "memory");   // DS RAW within wave

    Frag fr;
    const u16* pp = lp + (size_t)nl * 40 + hi * 8;
    fr.q[0] = *(const uint4v*)(pp);
    fr.q[1] = *(const uint4v*)(pp + 16);
    const v16bf aP = fr.v;

    // O += P (16x32) * V (32x64): V^T rows are contiguous in m
#pragma unroll
    for (int t = 0; t < 4; ++t) {
      v16bf bV = load_frag(Vh, 1024, t * 16, mc, lane);
      o[t] = wmma_bf16(aP, bV, o[t]);
    }
  }

  // normalize and write attn output as bf16 rows [b, n, h*64 + d]
#pragma unroll
  for (int r = 0; r < 8; ++r) {
    const float inv = 1.f / lsum[r];
    const int n = nTile + r + 8 * hi;
    u16* dst = attn + ((size_t)bb * 1024 + n) * 1024 + h * 64 + nl;
#pragma unroll
    for (int t = 0; t < 4; ++t) dst[t * 16] = f2bf(o[t][r] * inv);
  }
}

// ------------------------------------------------------------------ launch --
extern "C" void kernel_launch(void* const* d_in, const int* in_sizes, int n_in,
                              void* d_out, int out_size, void* d_ws, size_t ws_size,
                              hipStream_t stream) {
  const float* x    = (const float*)d_in[0];
  const float* ctx  = (const float*)d_in[1];
  const int*   mask = (const int*)  d_in[2];
  const float* Wq   = (const float*)d_in[3];
  const float* bq   = (const float*)d_in[4];
  const float* Wkv  = (const float*)d_in[5];
  const float* bkv  = (const float*)d_in[6];
  const float* Wp   = (const float*)d_in[7];
  const float* bp   = (const float*)d_in[8];
  float* out = (float*)d_out;

  char* ws = (char*)d_ws;
  u16* xbf  = (u16*)(ws);                          // 4M bf16 =  8MB
  u16* cbf  = (u16*)(ws + (size_t)( 8u << 20));    // 4M
  u16* Wqt  = (u16*)(ws + (size_t)(16u << 20));    // 1M
  u16* Wkvt = (u16*)(ws + (size_t)(18u << 20));    // 2M
  u16* Wpt  = (u16*)(ws + (size_t)(22u << 20));    // 1M
  u16* Qbf  = (u16*)(ws + (size_t)(24u << 20));    // 4M  [b,h,n,d]
  u16* Kbf  = (u16*)(ws + (size_t)(32u << 20));    // 4M  [b,h,m,d]
  u16* Vtb  = (u16*)(ws + (size_t)(40u << 20));    // 4M  [b,h,d,m]
  u16* attn = (u16*)(ws + (size_t)(48u << 20));    // 4M  [b,n,h*d]
  unsigned* mbits = (unsigned*)(ws + (size_t)(56u << 20));  // 128K words

  const int NACT = 4 * 1024 * 1024;
  convert_bf16<<<NACT / 256, 256, 0, stream>>>(x,   xbf, NACT);
  convert_bf16<<<NACT / 256, 256, 0, stream>>>(ctx, cbf, NACT);

  dim3 tb(32, 8);
  transpose_f32_to_bf16<<<dim3(32, 32), tb, 0, stream>>>(Wq,  Wqt,  1024, 1024);
  transpose_f32_to_bf16<<<dim3(64, 32), tb, 0, stream>>>(Wkv, Wkvt, 1024, 2048);
  transpose_f32_to_bf16<<<dim3(32, 32), tb, 0, stream>>>(Wp,  Wpt,  1024, 1024);

  const int NWORDS = 4 * 1024 * 32;                // 131072 mask words
  pack_mask<<<(NWORDS * 32) / 256, 256, 0, stream>>>(mask, mbits, NWORDS);

  // Q projection: 4096 x 1024 -> (4096/32)*(1024/32) = 4096 waves / 4 per block
  gemm_wmma<0><<<1024, 128, 0, stream>>>(xbf, Wqt, bq, Qbf, nullptr, nullptr,
                                         1024, 1024);
  // KV projection: 4096 x 2048 -> 8192 waves
  gemm_wmma<1><<<2048, 128, 0, stream>>>(cbf, Wkvt, bkv, Kbf, Vtb, nullptr,
                                         1024, 2048);
  // attention: 4*16*64 = 4096 waves / 4 per block
  flash_attn<<<1024, 128, 0, stream>>>(Qbf, Kbf, Vtb, mbits, attn);
  // output projection: 4096 x 1024, f32 out + bias
  gemm_wmma<2><<<1024, 128, 0, stream>>>(attn, Wpt, bp, nullptr, nullptr, out,
                                         1024, 1024);
}
